// self_att_layer_68753836474512
// MI455X (gfx1250) — matrix operand order repaired and checked by reading, not compiled, not run on hardware
//
#include <hip/hip_runtime.h>
#include <hip/hip_bf16.h>
#include <stdint.h>

#define N_NODES 8192
#define IN_CH   256
#define NEG_SLOPE 0.01f
#define WAVES_PER_WG 4

typedef __attribute__((ext_vector_type(16))) __bf16       v16bf;
typedef __attribute__((ext_vector_type(8)))  float        v8f;
typedef __attribute__((ext_vector_type(8)))  unsigned int v8u;

__device__ __forceinline__ uint32_t pk_bf16(float lo, float hi) {
  uint32_t ul = __builtin_bit_cast(uint32_t, lo);
  uint32_t uh = __builtin_bit_cast(uint32_t, hi);
  ul = (ul + 0x7FFFu + ((ul >> 16) & 1u)) >> 16;   // round-to-nearest-even
  uh = (uh + 0x7FFFu + ((uh >> 16) & 1u)) >> 16;
  return (ul & 0xFFFFu) | (uh << 16);
}

__device__ __forceinline__ float leaky01(float x) {
  return fmaxf(x, 0.f) + NEG_SLOPE * fminf(x, 0.f);
}

// K1: a = w1 @ W, b = w2 @ W   (HID == IN_CH == 256)
__global__ void k_proj_vec(const float* __restrict__ W,
                           const float* __restrict__ w1,
                           const float* __restrict__ w2,
                           float* __restrict__ avec, float* __restrict__ bvec) {
  int k = threadIdx.x;
  float sa = 0.f, sb = 0.f;
  for (int h = 0; h < IN_CH; ++h) {
    float w = W[h * IN_CH + k];
    sa += w1[h] * w;
    sb += w2[h] * w;
  }
  avec[k] = sa;
  bvec[k] = sb;
}

// K2: f1[i]=dot(a,x_i)+b1, f2[i]=dot(b,x_i)+b2, block-partial max of f2
__global__ void k_f12(const float* __restrict__ inlayer,
                      const float* __restrict__ avec, const float* __restrict__ bvec,
                      const float* __restrict__ b1p, const float* __restrict__ b2p,
                      float* __restrict__ f1, float* __restrict__ f2,
                      float* __restrict__ partial) {
  int i = blockIdx.x * blockDim.x + threadIdx.x;
  const float4* row = (const float4*)(inlayer + (size_t)i * IN_CH);
  const float4* av  = (const float4*)avec;
  const float4* bv  = (const float4*)bvec;
  float sa = 0.f, sb = 0.f;
#pragma unroll 8
  for (int t = 0; t < IN_CH / 4; ++t) {
    float4 x = row[t], A = av[t], B = bv[t];
    sa += x.x * A.x + x.y * A.y + x.z * A.z + x.w * A.w;
    sb += x.x * B.x + x.y * B.y + x.z * B.z + x.w * B.w;
  }
  sa += b1p[0];
  sb += b2p[0];
  f1[i] = sa;
  f2[i] = sb;

  __shared__ float red[256];
  red[threadIdx.x] = sb;
  __syncthreads();
  for (int s = 128; s > 0; s >>= 1) {
    if (threadIdx.x < s) red[threadIdx.x] = fmaxf(red[threadIdx.x], red[threadIdx.x + s]);
    __syncthreads();
  }
  if (threadIdx.x == 0) partial[blockIdx.x] = red[0];
}

// K3: reduce 32 block partials -> global max(f2)
__global__ void k_maxred(const float* __restrict__ partial, float* __restrict__ f2max) {
  float v = partial[threadIdx.x];
  for (int off = 16; off >= 1; off >>= 1) v = fmaxf(v, __shfl_xor(v, off, 32));
  if (threadIdx.x == 0) f2max[0] = v;
}

// K4: pre-swizzle inlayer (f32) into the WMMA-B per-lane bf16 layout.
// Tile (jt,nt): B[k][n] = inlayer[jt*32+k][nt*16+n], k=0..31, n=0..15.
// Lane l: n = l&15, kb = (l<16)?0:16; packed dword v holds K = kb+2v (lo), kb+2v+1 (hi).
// Storage: bprep[((jt*16+nt)*32 + lane)*8 .. +7]  (32B per lane, 16B-aligned)
__global__ void k_bprep(const float* __restrict__ inlayer, uint32_t* __restrict__ bprep) {
  int id   = blockIdx.x * blockDim.x + threadIdx.x;  // 256 jt * 16 nt * 32 lanes = 131072
  int lane = id & 31;
  int nt   = (id >> 5) & 15;
  int jt   = id >> 9;
  int j0   = jt * 32;
  int kb   = (lane < 16) ? 0 : 16;
  int col  = nt * 16 + (lane & 15);
  uint32_t pk[8];
#pragma unroll
  for (int v = 0; v < 8; ++v) {
    float lo = inlayer[(size_t)(j0 + kb + 2 * v)     * IN_CH + col];
    float hi = inlayer[(size_t)(j0 + kb + 2 * v + 1) * IN_CH + col];
    pk[v] = pk_bf16(lo, hi);
  }
  uint4* dst = (uint4*)(bprep + (size_t)id * 8);
  dst[0] = make_uint4(pk[0], pk[1], pk[2], pk[3]);
  dst[1] = make_uint4(pk[4], pk[5], pk[6], pk[7]);
}

// K5: fused masked-softmax-attention GEMM.
// Wave owns 16 rows; streams adj row tiles (32 wide), builds p=exp(leaky(f1+f2)-c)
// directly in the WMMA-A bf16 register layout, and accumulates p @ inlayer_bf16
// with 16x v_wmma_f32_16x16x32_bf16 per k-tile (full 256-wide output resident
// in 128 accumulation VGPRs). Row sums ride along in f32; final store scales by 1/sum.
__global__ void __launch_bounds__(32 * WAVES_PER_WG)
k_gat_main(const int* __restrict__ adj,
           const uint32_t* __restrict__ bprep,
           const float* __restrict__ f1,
           const float* __restrict__ f2,
           const float* __restrict__ f2max,
           float* __restrict__ out) {
  const int lane   = threadIdx.x & 31;
  const int wave   = threadIdx.x >> 5;
  const int rowset = blockIdx.x * WAVES_PER_WG + wave;   // 0..511
  const int i0     = rowset * 16;
  const int mrow   = lane & 15;
  const int half   = lane >> 4;                          // 0: K 0..15, 1: K 16..31
  const int kb     = half ? 16 : 0;
  const int iA     = i0 + mrow;

  const float f1i = f1[iA];
  const float ci  = leaky01(f1i + f2max[0]);             // safe upper bound of row max
  const int* adjrow = adj + (size_t)iA * N_NODES;
  const v8u* bpv = (const v8u*)bprep;

  v8f C[16] = {};
  float ssum = 0.f;

  for (int jt = 0; jt < N_NODES / 32; ++jt) {
    const int j0 = jt * 32;
    __builtin_prefetch(adjrow + j0 + 512, 0, 1);         // global_prefetch ahead on adj stream

    const int4*   ap0 = (const int4*)(adjrow + j0 + kb);
    const int4*   ap1 = (const int4*)(adjrow + j0 + kb + 8);
    const float4* fp0 = (const float4*)(f2 + j0 + kb);
    const float4* fp1 = (const float4*)(f2 + j0 + kb + 8);
    int4   a0 = ap0[0], a1 = ap0[1], a2 = ap1[0], a3 = ap1[1];
    float4 g0 = fp0[0], g1 = fp0[1], g2 = fp1[0], g3 = fp1[1];

    const int   am[16] = {a0.x, a0.y, a0.z, a0.w, a1.x, a1.y, a1.z, a1.w,
                          a2.x, a2.y, a2.z, a2.w, a3.x, a3.y, a3.z, a3.w};
    const float gm[16] = {g0.x, g0.y, g0.z, g0.w, g1.x, g1.y, g1.z, g1.w,
                          g2.x, g2.y, g2.z, g2.w, g3.x, g3.y, g3.z, g3.w};

    float p[16];
#pragma unroll
    for (int e = 0; e < 16; ++e) {
      float t  = f1i + gm[e];
      float lv = fmaxf(t, 0.f) + NEG_SLOPE * fminf(t, 0.f);
      float pe = (am[e] > 0) ? __expf(lv - ci) : 0.f;    // mask == -1e9 bias
      ssum += pe;
      p[e] = pe;
    }

    v8u au;
#pragma unroll
    for (int v = 0; v < 8; ++v) au[v] = pk_bf16(p[2 * v], p[2 * v + 1]);
    v16bf A = __builtin_bit_cast(v16bf, au);

    const size_t bbase = (size_t)(jt * 16) * 32 + lane;
#pragma unroll
    for (int nt = 0; nt < 16; ++nt) {
      v8u bu = bpv[bbase + (size_t)nt * 32];             // 2x global_load_b128
      v16bf B = __builtin_bit_cast(v16bf, bu);
      C[nt] = __builtin_amdgcn_wmma_f32_16x16x32_bf16(
          false, A, false, B, (short)0, C[nt], false, false);
    }
  }

  // Row sums: lane l and l+16 cover disjoint K slices of row i0+(l&15).
  float srow = ssum + __shfl_xor(ssum, 16, 32);
  float inv[8];
#pragma unroll
  for (int r = 0; r < 8; ++r) {
    float s = __shfl(srow, r + half * 8, 32);            // lanes 0..15 hold rows 0..15
    inv[r] = 1.f / (s + 1e-30f);
  }

  // C VGPR r -> row r + 8*half, col = mrow (ISA C/D layout)
#pragma unroll
  for (int nt = 0; nt < 16; ++nt) {
#pragma unroll
    for (int r = 0; r < 8; ++r) {
      out[(size_t)(i0 + r + half * 8) * IN_CH + nt * 16 + mrow] = C[nt][r] * inv[r];
    }
  }
}

extern "C" void kernel_launch(void* const* d_in, const int* in_sizes, int n_in,
                              void* d_out, int out_size, void* d_ws, size_t ws_size,
                              hipStream_t stream) {
  const float* inlayer = (const float*)d_in[0];
  const int*   adj     = (const int*)d_in[1];
  const float* W       = (const float*)d_in[2];
  const float* w1      = (const float*)d_in[3];
  const float* b1      = (const float*)d_in[4];
  const float* w2      = (const float*)d_in[5];
  const float* b2      = (const float*)d_in[6];
  float* out = (float*)d_out;

  char* ws = (char*)d_ws;
  float*    avec    = (float*)(ws);                      // 1 KB
  float*    bvec    = (float*)(ws + 1024);               // 1 KB
  float*    f1      = (float*)(ws + 2048);               // 32 KB
  float*    f2      = (float*)(ws + 2048 + 32768);       // 32 KB
  float*    partial = (float*)(ws + 2048 + 65536);       // 128 B
  float*    f2max   = (float*)(ws + 2048 + 65536 + 256); // 4 B
  uint32_t* bprep   = (uint32_t*)(ws + (1 << 17));       // 4 MB bf16 B-layout

  k_proj_vec<<<1, 256, 0, stream>>>(W, w1, w2, avec, bvec);
  k_f12<<<N_NODES / 256, 256, 0, stream>>>(inlayer, avec, bvec, b1, b2, f1, f2, partial);
  k_maxred<<<1, 32, 0, stream>>>(partial, f2max);
  k_bprep<<<(N_NODES / 32) * 16 * 32 / 256, 256, 0, stream>>>(inlayer, bprep);
  k_gat_main<<<(N_NODES / 16) / WAVES_PER_WG, 32 * WAVES_PER_WG, 0, stream>>>(
      adj, bprep, f1, f2, f2max, out);
}
